// Lin_Transpose_NonSquare_85641647882598
// MI455X (gfx1250) — matrix operand order repaired and checked by reading, not compiled, not run on hardware
//
#include <hip/hip_runtime.h>

// ---------------------------------------------------------------------------
// out[b,s,o] = x @ Weff where Weff = sum of 13 permuted copies of W^T.
// Strategy: build Weff^T once (bf16 hi/lo split), split x (bf16 hi/lo),
// then a bf16x3 split-precision WMMA GEMM (hi*hi + hi*lo + lo*hi) which
// recovers ~f32 accuracy at bf16 matrix-core rates. All operands are
// L2-resident on MI455X (x:64MB + W:32MB << 192MB L2) -> compute bound.
// ---------------------------------------------------------------------------

typedef __attribute__((ext_vector_type(16))) __bf16 v16bf;
typedef __attribute__((ext_vector_type(8)))  float  v8f;
typedef __attribute__((ext_vector_type(4)))  __bf16 bf16x4;

#define IN_F   2048
#define OUT_F  4096
#define MTOT   8192               // 4 * 2048 rows of x
#define BM     128
#define BN     128
#define BK     32
#define KTILES (IN_F / BK)        // 64

// ---------------- stage 1: split x (f32) into bf16 hi + lo ----------------
__global__ __launch_bounds__(256) void split_x_kernel(
    const float* __restrict__ x, __bf16* __restrict__ xhi, __bf16* __restrict__ xlo) {
  const long i = ((long)blockIdx.x * 256 + threadIdx.x) * 4;
  const float4 v = *(const float4*)(x + i);
  bf16x4 h, l;
  h.x = (__bf16)v.x; l.x = (__bf16)(v.x - (float)h.x);
  h.y = (__bf16)v.y; l.y = (__bf16)(v.y - (float)h.y);
  h.z = (__bf16)v.z; l.z = (__bf16)(v.z - (float)h.z);
  h.w = (__bf16)v.w; l.w = (__bf16)(v.w - (float)h.w);
  *(bf16x4*)(xhi + i) = h;
  *(bf16x4*)(xlo + i) = l;
}

// -------- stage 2: Weff^T[o][i] = sum of 13 permuted reads of W[o][i] ------
// Stored N-major, K-contiguous so it loads directly as the WMMA B operand.
__global__ __launch_bounds__(256) void build_weff_kernel(
    const float* __restrict__ W, __bf16* __restrict__ whi, __bf16* __restrict__ wlo) {
  const long t = (long)blockIdx.x * 256 + threadIdx.x;
  const int  o = (int)(t >> 11);          // / 2048
  const int  i = (int)(t & (IN_F - 1));
  // Weff[i,o] = W[o,i] + W[o,2047-i] + W[4095-o,i]
  //           + sum_{s=1..5} W[(o-s)%4096, i] + W[o, (i-s)%2048]
  float s = W[(long)o * IN_F + i]
          + W[(long)o * IN_F + (IN_F - 1 - i)]
          + W[(long)(OUT_F - 1 - o) * IN_F + i];
#pragma unroll
  for (int sh = 1; sh <= 5; ++sh) {
    s += W[(long)((o - sh) & (OUT_F - 1)) * IN_F + i];
    s += W[(long)o * IN_F + ((i - sh) & (IN_F - 1))];
  }
  const __bf16 h = (__bf16)s;
  whi[t] = h;
  wlo[t] = (__bf16)(s - (float)h);
}

// ---------------- stage 3: bf16x3 WMMA GEMM, 128x128 tiles -----------------
__global__ __launch_bounds__(256) void gemm_bf16x3_kernel(
    const __bf16* __restrict__ xhi, const __bf16* __restrict__ xlo,
    const __bf16* __restrict__ whi, const __bf16* __restrict__ wlo,
    float* __restrict__ out) {
  // double-buffered tiles: 2 bufs * (hi+lo) * 128*32 bf16 = 32KB A + 32KB B
  __shared__ __align__(16) __bf16 sA[2][2][BM * BK];
  __shared__ __align__(16) __bf16 sB[2][2][BN * BK];

  const int t    = threadIdx.x;
  const int lane = t & 31;            // wave32
  const int wid  = t >> 5;            // 8 waves
  const int wm   = (wid >> 2) * 64;   // wave M offset inside tile
  const int wn   = (wid & 3) * 32;    // wave N offset inside tile
  const int lr   = lane & 15;
  const int kbA  = (lane >> 4) * 8;   // A frag: K base 0/8 (chunks +0,+16)
  const int kbB  = (lane >> 4) * 8;   // B frag: K base 0/8 (chunks +0,+16)

  const long tileM = (long)blockIdx.x * BM;
  const long tileN = (long)blockIdx.y * BN;

  // global->LDS staging: 512 16B chunks per array; thread owns rows r, r+64
  const int row0 = t >> 2;
  const int row1 = row0 + 64;
  const int c0   = (t & 3) * 8;       // bf16 element offset of 16B chunk
  const long rstep = 64L * IN_F;

  const __bf16* pAh = xhi + (tileM + row0) * (long)IN_F + c0;
  const __bf16* pAl = xlo + (tileM + row0) * (long)IN_F + c0;
  const __bf16* pBh = whi + (tileN + row0) * (long)IN_F + c0;
  const __bf16* pBl = wlo + (tileN + row0) * (long)IN_F + c0;

#define LD4(p) (*(const uint4*)(p))

  v8f acc[4][2] = {};

  union Frag { uint4 u[2]; v16bf v; };

  auto compute = [&](int buf) {
    Frag bh[2], bl[2];
#pragma unroll
    for (int nt = 0; nt < 2; ++nt) {
      const __bf16* pb  = &sB[buf][0][(wn + nt * 16 + lr) * BK + kbB];
      const __bf16* pbl = &sB[buf][1][(wn + nt * 16 + lr) * BK + kbB];
      bh[nt].u[0] = LD4(pb);      bh[nt].u[1] = LD4(pb + 16);
      bl[nt].u[0] = LD4(pbl);     bl[nt].u[1] = LD4(pbl + 16);
    }
#pragma unroll
    for (int mt = 0; mt < 4; ++mt) {
      Frag ah, al;
      const __bf16* pa  = &sA[buf][0][(wm + mt * 16 + lr) * BK + kbA];
      const __bf16* pal = &sA[buf][1][(wm + mt * 16 + lr) * BK + kbA];
      ah.u[0] = LD4(pa);   ah.u[1] = LD4(pa + 16);
      al.u[0] = LD4(pal);  al.u[1] = LD4(pal + 16);
#pragma unroll
      for (int nt = 0; nt < 2; ++nt) {
        acc[mt][nt] = __builtin_amdgcn_wmma_f32_16x16x32_bf16(
            false, ah.v, false, bh[nt].v, (short)0, acc[mt][nt], false, false);
        acc[mt][nt] = __builtin_amdgcn_wmma_f32_16x16x32_bf16(
            false, ah.v, false, bl[nt].v, (short)0, acc[mt][nt], false, false);
        acc[mt][nt] = __builtin_amdgcn_wmma_f32_16x16x32_bf16(
            false, al.v, false, bh[nt].v, (short)0, acc[mt][nt], false, false);
      }
    }
  };

  // preload k-tile 0 into buffer 0
  {
    *(uint4*)&sA[0][0][row0 * BK + c0] = LD4(pAh);
    *(uint4*)&sA[0][0][row1 * BK + c0] = LD4(pAh + rstep);
    *(uint4*)&sA[0][1][row0 * BK + c0] = LD4(pAl);
    *(uint4*)&sA[0][1][row1 * BK + c0] = LD4(pAl + rstep);
    *(uint4*)&sB[0][0][row0 * BK + c0] = LD4(pBh);
    *(uint4*)&sB[0][0][row1 * BK + c0] = LD4(pBh + rstep);
    *(uint4*)&sB[0][1][row0 * BK + c0] = LD4(pBl);
    *(uint4*)&sB[0][1][row1 * BK + c0] = LD4(pBl + rstep);
  }
  __syncthreads();

  for (int kt = 0; kt < KTILES; ++kt) {
    const int  buf = kt & 1;
    const bool pf  = (kt + 1 < KTILES);
    uint4 a0, a1, al0, al1, b0, b1, bl0, bl1;
    if (pf) {  // prefetch next k-tile to registers, overlapping the WMMAs
      const int off = (kt + 1) * BK;
      a0  = LD4(pAh + off);  a1  = LD4(pAh + rstep + off);
      al0 = LD4(pAl + off);  al1 = LD4(pAl + rstep + off);
      b0  = LD4(pBh + off);  b1  = LD4(pBh + rstep + off);
      bl0 = LD4(pBl + off);  bl1 = LD4(pBl + rstep + off);
    }

    compute(buf);

    if (pf) {
      const int nb = buf ^ 1;
      *(uint4*)&sA[nb][0][row0 * BK + c0] = a0;
      *(uint4*)&sA[nb][0][row1 * BK + c0] = a1;
      *(uint4*)&sA[nb][1][row0 * BK + c0] = al0;
      *(uint4*)&sA[nb][1][row1 * BK + c0] = al1;
      *(uint4*)&sB[nb][0][row0 * BK + c0] = b0;
      *(uint4*)&sB[nb][0][row1 * BK + c0] = b1;
      *(uint4*)&sB[nb][1][row0 * BK + c0] = bl0;
      *(uint4*)&sB[nb][1][row1 * BK + c0] = bl1;
      __syncthreads();
    }
  }

  // epilogue: 16x16 f32 C/D layout — lane = N, VGPR r = M (+8 for hi lanes)
#pragma unroll
  for (int mt = 0; mt < 4; ++mt) {
    const long m0 = tileM + wm + mt * 16 + (lane >> 4) * 8;
#pragma unroll
    for (int nt = 0; nt < 2; ++nt) {
      const long n = tileN + wn + nt * 16 + lr;
      float* p = out + m0 * OUT_F + n;
#pragma unroll
      for (int r = 0; r < 8; ++r) p[(long)r * OUT_F] = acc[mt][nt][r];
    }
  }
#undef LD4
}

// ---------------------------------------------------------------------------
extern "C" void kernel_launch(void* const* d_in, const int* in_sizes, int n_in,
                              void* d_out, int out_size, void* d_ws, size_t ws_size,
                              hipStream_t stream) {
  const float* x = (const float*)d_in[0];   // (4,2048,2048) f32
  const float* W = (const float*)d_in[1];   // (4096,2048)   f32
  float* out = (float*)d_out;               // (4,2048,4096) f32

  const size_t xElems = (size_t)MTOT * IN_F;   // 16.78M
  const size_t wElems = (size_t)OUT_F * IN_F;  //  8.39M
  // workspace layout (96 MB total): xhi | xlo | whi | wlo
  __bf16* xhi = (__bf16*)d_ws;
  __bf16* xlo = xhi + xElems;
  __bf16* whi = xlo + xElems;
  __bf16* wlo = whi + wElems;

  split_x_kernel<<<(unsigned)(xElems / 4 / 256), 256, 0, stream>>>(x, xhi, xlo);
  build_weff_kernel<<<(unsigned)(wElems / 256), 256, 0, stream>>>(W, whi, wlo);

  dim3 grid(MTOT / BM, OUT_F / BN);  // 64 x 32 workgroups
  gemm_bf16x3_kernel<<<grid, 256, 0, stream>>>(xhi, xlo, whi, wlo, out);
}
